// CrossAttentionClusteringLearnableK_70798240907309
// MI455X (gfx1250) — matrix operand order repaired and checked by reading, not compiled, not run on hardware
//
#include <hip/hip_runtime.h>
#include <hip/hip_bf16.h>
#include <stdint.h>

#ifndef __has_builtin
#define __has_builtin(x) 0
#endif

#if __has_builtin(__builtin_amdgcn_global_load_async_to_lds_b128)
#define HAVE_ASYNC_LDS 1
#else
#define HAVE_ASYNC_LDS 0
#endif

#define H_    8
#define DH_   64
#define DQ_   256
#define DHIT_ 256
#define DMLP_ 512
#define B_    8
#define K_    64
#define N_    16384
#define TN    32
#define G_    8                 // split-N groups per (b,h) -> 512 workgroups
#define NG_   (N_ / G_)         // 2048 hits per group

typedef __bf16 bf16_t;
typedef __attribute__((ext_vector_type(16))) __bf16 v16bf;
typedef __attribute__((ext_vector_type(8)))  __bf16 v8bf;
typedef __attribute__((ext_vector_type(4)))  __bf16 v4bf;
typedef __attribute__((ext_vector_type(8)))  float  v8f;
typedef __attribute__((ext_vector_type(4)))  int    v4i;

#if HAVE_ASYNC_LDS
typedef __attribute__((address_space(1))) v4i* as1_v4i_p;   // global int4*
typedef __attribute__((address_space(3))) v4i* as3_v4i_p;   // LDS int4*
#endif

// Native conversion: lowers to v_cvt_pk_bf16_f32 pairs on gfx1250 (RNE).
static __device__ __forceinline__ bf16_t f2bf(float f) { return (bf16_t)f; }

static __device__ __forceinline__ v16bf cat16(v8bf lo, v8bf hi) {
  return __builtin_shufflevector(lo, hi, 0,1,2,3,4,5,6,7,8,9,10,11,12,13,14,15);
}

static __device__ __forceinline__ v8f wmma_bf16(v16bf a, v16bf b, v8f c) {
  return __builtin_amdgcn_wmma_f32_16x16x32_bf16(false, a, false, b, (short)0, c,
                                                 false, false);
}

// A fragment (16x32 bf16) from a row-major LDS buffer, contiguous K order:
// lane<16: K=0..15; lane>=16: K=16..31.
static __device__ __forceinline__ v16bf load_a_frag(const bf16_t* buf, int pitch,
                                                    int mrow, int kb, int lane) {
  const bf16_t* p = buf + (mrow + (lane & 15)) * pitch + kb * 32 + ((lane >> 4) & 1) * 16;
  v8bf lo = *(const v8bf*)p;
  v8bf hi = *(const v8bf*)(p + 8);
  return cat16(lo, hi);
}

static __device__ __forceinline__ float block_sum(float v, float* red) {
#pragma unroll
  for (int o = 16; o > 0; o >>= 1) v += __shfl_xor(v, o, 32);
  const int wave = threadIdx.x >> 5, lane = threadIdx.x & 31;
  __syncthreads();
  if (lane == 0) red[wave] = v;
  __syncthreads();
  float tot = 0.f;
#pragma unroll
  for (int i = 0; i < 8; ++i) tot += red[i];
  return tot;
}

#if HAVE_ASYNC_LDS
static __device__ __forceinline__ void wait_async0() {
#if __has_builtin(__builtin_amdgcn_s_wait_asynccnt)
  __builtin_amdgcn_s_wait_asynccnt(0);
#else
  asm volatile("s_wait_asynccnt 0" ::: "memory");
#endif
}
#endif

// ---------------------------------------------------------------------------
// Kernel 0: pack Wk/Wv [256][512] f32 -> bf16 WMMA B-fragments.
// Fragment (h, ct, kb): 32x16 bf16, stored lane-major, 16 contiguous per lane.
// Lane l holds column n = l%16; K = kb*32 + (l/16)*16 + j (j=0..15).
__global__ __launch_bounds__(256) void pack_w_kernel(const float* __restrict__ W,
                                                     bf16_t* __restrict__ out) {
  const int gid  = blockIdx.x * 256 + threadIdx.x;   // 0..8191
  const int lane = gid & 31;
  const int f    = gid >> 5;                         // (h, ct, kb)
  const int kb   = f & 7;
  const int ct   = (f >> 3) & 3;
  const int h    = f >> 5;
  const int col   = h * 64 + ct * 16 + (lane & 15);
  const int rbase = kb * 32 + ((lane >> 4) & 1) * 16;
  bf16_t* dst = out + (size_t)gid * 16;
#pragma unroll
  for (int j = 0; j < 16; ++j)
    dst[j] = f2bf(W[(size_t)(rbase + j) * 512 + col]);
}

// ---------------------------------------------------------------------------
// Kernel 1: LayerNorm(slots) @ Wq, scale 1/sqrt(DH) folded in, bf16 out.
// q layout: [B][H][K][DH]
__global__ __launch_bounds__(256) void ln_q_kernel(const float* __restrict__ slots,
                                                   const float* __restrict__ g,
                                                   const float* __restrict__ beta,
                                                   const float* __restrict__ Wq,
                                                   bf16_t* __restrict__ qout) {
  const int row = blockIdx.x;               // b*64 + k
  const int tid = threadIdx.x;
  __shared__ float sn[DQ_];
  __shared__ float red[8];
  const float x = slots[(size_t)row * DQ_ + tid];
  const float mean = block_sum(x, red) * (1.f / DQ_);
  const float d = x - mean;
  const float var = block_sum(d * d, red) * (1.f / DQ_);
  sn[tid] = d * rsqrtf(var + 1e-5f) * g[tid] + beta[tid];
  __syncthreads();
  const int b = row >> 6, k = row & 63;
  for (int c = tid; c < H_ * DH_; c += 256) {
    float acc = 0.f;
    for (int j = 0; j < DQ_; ++j) acc = fmaf(sn[j], Wq[(size_t)j * (H_ * DH_) + c], acc);
    const int h = c >> 6, dh = c & 63;
    qout[((((size_t)b * H_ + h) * K_ + k) * DH_) + dh] = f2bf(acc * 0.125f);
  }
}

// ---------------------------------------------------------------------------
// Kernel 2: fused K/V projection + flash attention, split-N.
// One workgroup per (b, h, group); group g handles hits [g*NG_, (g+1)*NG_).
// Emits unnormalized partial O plus (m, l) stats per slot row.
__global__ __launch_bounds__(256) void attn_kernel(const float* __restrict__ hits,
                                                   const bf16_t* __restrict__ qg,
                                                   const bf16_t* __restrict__ wkb,
                                                   const bf16_t* __restrict__ wvb,
                                                   float* __restrict__ po,
                                                   float* __restrict__ mst,
                                                   float* __restrict__ lst) {
  const int gid = blockIdx.x;                // 0..511
  const int grp = gid & (G_ - 1);
  const int bh  = gid >> 3;                  // 0..63
  const int b = bh >> 3, h = bh & 7;
  const int tid = threadIdx.x;
  const int wave = tid >> 5, lane = tid & 31;

#if HAVE_ASYNC_LDS
  __shared__ float  stage_s[TN * DHIT_];     // 32 KB raw f32 staging (async dest)
#endif
  __shared__ bf16_t hits_s[TN * DHIT_];      // 16 KB hit tile (bf16)
  __shared__ bf16_t q_s[K_ * DH_];           //  8 KB Q tile
  __shared__ bf16_t k_s[TN * DH_];           //  4 KB [hit][dh]
  __shared__ bf16_t vT_s[DH_ * TN];          //  4 KB [dv][hit]
  __shared__ float  s_s[K_ * TN];            //  8 KB logits
  __shared__ bf16_t p_s[K_ * TN];            //  4 KB exp(S - m)
  __shared__ float  m_st[K_], l_st[K_], a_st[K_];

  {
    const v8bf* src = (const v8bf*)(qg + (size_t)bh * (K_ * DH_));
    v8bf* dst = (v8bf*)q_s;
    dst[tid] = src[tid];
    dst[tid + 256] = src[tid + 256];
  }
  if (tid < K_) { m_st[tid] = -3.0e38f; l_st[tid] = 0.f; }

  v8f o0 = {}; v8f o1 = {};
  const int t0 = wave * 2, t1 = wave * 2 + 1;       // O tiles (4x4 grid of 16x16)
  const int rtO = t0 >> 2;                          // shared row-tile
  const int mO = rtO * 16 + 8 * (lane >> 4);
  const int c0 = (t0 & 3) * 16 + (lane & 15);
  const int c1 = (t1 & 3) * 16 + (lane & 15);

  const float* hb = hits + (size_t)b * N_ * DHIT_;
  const int nbeg = grp * NG_;
  const int nend = nbeg + NG_;

#if HAVE_ASYNC_LDS
  // prologue: kick off DMA of this group's first tile into LDS staging
  {
    const float4* src = (const float4*)(hb + (size_t)nbeg * DHIT_);
#pragma unroll
    for (int i = 0; i < 8; ++i) {
      const int idx = tid + i * 256;                // 2048 float4 per tile
      __builtin_amdgcn_global_load_async_to_lds_b128(
          (as1_v4i_p)(src + idx), (as3_v4i_p)(stage_s + idx * 4), 0, 0);
    }
  }
#endif

  for (int n0 = nbeg; n0 < nend; n0 += TN) {
#if HAVE_ASYNC_LDS
    wait_async0();                           // this wave's DMA for tile n0 done
    __syncthreads();                         // ... visible to all waves
    // convert staged f32 tile -> bf16 hit tile
    {
#pragma unroll
      for (int i = 0; i < 8; ++i) {
        const int idx = tid + i * 256;
        const float4 f = ((const float4*)stage_s)[idx];
        v4bf pk;
        pk[0] = f2bf(f.x); pk[1] = f2bf(f.y); pk[2] = f2bf(f.z); pk[3] = f2bf(f.w);
        *(v4bf*)&hits_s[idx * 4] = pk;
      }
    }
    __syncthreads();                         // stage reads done; hits_s ready
    if (n0 + TN < nend) {                    // overlap next tile's DMA with compute
      const float4* src = (const float4*)(hb + (size_t)(n0 + TN) * DHIT_);
#pragma unroll
      for (int i = 0; i < 8; ++i) {
        const int idx = tid + i * 256;
        __builtin_amdgcn_global_load_async_to_lds_b128(
            (as1_v4i_p)(src + idx), (as3_v4i_p)(stage_s + idx * 4), 0, 0);
      }
    }
#else
    __syncthreads();
    // synchronous staging fallback: global f32 -> bf16 LDS through VGPRs
    {
      const float4* src = (const float4*)(hb + (size_t)n0 * DHIT_);
#pragma unroll
      for (int i = 0; i < 8; ++i) {
        const int idx = tid + i * 256;
        const float4 f = src[idx];
        v4bf pk;
        pk[0] = f2bf(f.x); pk[1] = f2bf(f.y); pk[2] = f2bf(f.z); pk[3] = f2bf(f.w);
        *(v4bf*)&hits_s[idx * 4] = pk;
      }
      if (n0 + TN < nend)
        __builtin_prefetch((const char*)(hb + (size_t)(n0 + TN) * DHIT_) + tid * 128, 0, 1);
    }
    __syncthreads();
#endif
    // 2) K/V projection: wave -> one 16x16 tile of K and of V (8 k-steps each)
    {
      const int rt = wave >> 2, ct = wave & 3;
      v8f ck = {}; v8f cv = {};
#pragma unroll
      for (int kb = 0; kb < 8; ++kb) {
        const v16bf a = load_a_frag(hits_s, DHIT_, rt * 16, kb, lane);
        const size_t foff = ((size_t)((h * 4 + ct) * 8 + kb) * 512) + (size_t)lane * 16;
        const v16bf bk = *(const v16bf*)(wkb + foff);
        const v16bf bv = *(const v16bf*)(wvb + foff);
        ck = wmma_bf16(a, bk, ck);
        cv = wmma_bf16(a, bv, cv);
      }
      const int mrow = rt * 16 + 8 * (lane >> 4);
      const int ncol = ct * 16 + (lane & 15);
#pragma unroll
      for (int r = 0; r < 8; ++r) k_s[(mrow + r) * DH_ + ncol] = f2bf(ck[r]);
      v8bf pv;
#pragma unroll
      for (int r = 0; r < 8; ++r) pv[r] = f2bf(cv[r]);
      *(v8bf*)&vT_s[ncol * TN + mrow] = pv;         // transposed, contiguous store
    }
    __syncthreads();
    // 3) S = Q @ K^T (64x32 logits; scale pre-folded into Q)
    {
      const int rt = wave >> 1, ct = wave & 1;
      v8f cs = {};
#pragma unroll
      for (int kb = 0; kb < 2; ++kb) {
        const v16bf a = load_a_frag(q_s, DH_, rt * 16, kb, lane);
        const bf16_t* rp = &k_s[(ct * 16 + (lane & 15)) * DH_ + kb * 32 + (lane >> 4) * 16];
        const v16bf bfrag = cat16(*(const v8bf*)rp, *(const v8bf*)(rp + 8));
        cs = wmma_bf16(a, bfrag, cs);
      }
      const int mrow = rt * 16 + 8 * (lane >> 4);
      const int ncol = ct * 16 + (lane & 15);
#pragma unroll
      for (int r = 0; r < 8; ++r) s_s[(mrow + r) * TN + ncol] = cs[r];
    }
    __syncthreads();
    // 4) online softmax update (4 lanes per row, wave-local reductions)
    {
      const int row = tid >> 2, seg = tid & 3;
      const float4* sr4 = (const float4*)&s_s[row * TN + seg * 8];
      const float4 a0 = sr4[0], a1 = sr4[1];
      float sv[8] = {a0.x, a0.y, a0.z, a0.w, a1.x, a1.y, a1.z, a1.w};
      float mx = -3.0e38f;
#pragma unroll
      for (int j = 0; j < 8; ++j) mx = fmaxf(mx, sv[j]);
      mx = fmaxf(mx, __shfl_xor(mx, 1, 32));
      mx = fmaxf(mx, __shfl_xor(mx, 2, 32));
      const float mold = m_st[row];
      const float mnew = fmaxf(mold, mx);
      float ps = 0.f;
      v4bf pk0, pk1;
#pragma unroll
      for (int j = 0; j < 8; ++j) {
        const float p = __expf(sv[j] - mnew);
        ps += p;
        if (j < 4) pk0[j] = f2bf(p); else pk1[j - 4] = f2bf(p);
      }
      *(v4bf*)&p_s[row * TN + seg * 8]     = pk0;
      *(v4bf*)&p_s[row * TN + seg * 8 + 4] = pk1;
      ps += __shfl_xor(ps, 1, 32);
      ps += __shfl_xor(ps, 2, 32);
      if (seg == 0) {
        const float al = __expf(mold - mnew);
        a_st[row] = al;
        m_st[row] = mnew;
        l_st[row] = l_st[row] * al + ps;
      }
    }
    __syncthreads();
    // 5) rescale O then O += P @ V
    {
#pragma unroll
      for (int r = 0; r < 8; ++r) {
        const float al = a_st[mO + r];
        o0[r] *= al;
        o1[r] *= al;
      }
      const v16bf a = load_a_frag(p_s, TN, rtO * 16, 0, lane);
      const bf16_t* r0 = &vT_s[c0 * TN + (lane >> 4) * 16];
      const bf16_t* r1 = &vT_s[c1 * TN + (lane >> 4) * 16];
      const v16bf bv0 = cat16(*(const v8bf*)r0, *(const v8bf*)(r0 + 8));
      const v16bf bv1 = cat16(*(const v8bf*)r1, *(const v8bf*)(r1 + 8));
      o0 = wmma_bf16(a, bv0, o0);
      o1 = wmma_bf16(a, bv1, o1);
    }
  }
  __syncthreads();
  // epilogue: write unnormalized partial O + (m, l) stats for this group
  {
    float* pob = po + (size_t)gid * (K_ * DH_);     // [gid][64][64]
#pragma unroll
    for (int r = 0; r < 8; ++r) {
      pob[(size_t)(mO + r) * DH_ + c0] = o0[r];
      pob[(size_t)(mO + r) * DH_ + c1] = o1[r];
    }
    if (tid < K_) {
      mst[(size_t)gid * K_ + tid] = m_st[tid];
      lst[(size_t)gid * K_ + tid] = l_st[tid];
    }
  }
}

// ---------------------------------------------------------------------------
// Kernel 2b: merge split-N partials. One workgroup per (b,h).
// O = sum_g exp(m_g - m*) O_g ; l = sum_g exp(m_g - m*) l_g ; attv = O / l.
__global__ __launch_bounds__(256) void attn_reduce_kernel(const float* __restrict__ po,
                                                          const float* __restrict__ mst,
                                                          const float* __restrict__ lst,
                                                          float* __restrict__ attv) {
  const int bh = blockIdx.x;                 // 0..63
  const int b = bh >> 3, h = bh & 7;
  const int tid = threadIdx.x;
  __shared__ float alpha[K_ * G_];           // [row][g]
  __shared__ float linv[K_];
  if (tid < K_) {
    const int row = tid;
    float mg[G_], lg[G_];
    float mmax = -3.0e38f;
#pragma unroll
    for (int g = 0; g < G_; ++g) {
      mg[g] = mst[((size_t)bh * G_ + g) * K_ + row];
      lg[g] = lst[((size_t)bh * G_ + g) * K_ + row];
      mmax = fmaxf(mmax, mg[g]);
    }
    float ltot = 0.f;
#pragma unroll
    for (int g = 0; g < G_; ++g) {
      const float al = __expf(mg[g] - mmax);
      alpha[row * G_ + g] = al;
      ltot += al * lg[g];
    }
    linv[row] = 1.0f / ltot;
  }
  __syncthreads();
  for (int e = tid; e < K_ * DH_; e += 256) {
    const int row = e >> 6, col = e & 63;
    float acc = 0.f;
#pragma unroll
    for (int g = 0; g < G_; ++g)
      acc = fmaf(alpha[row * G_ + g],
                 po[((size_t)(bh * G_ + g) * K_ + row) * DH_ + col], acc);
    attv[((size_t)b * K_ + row) * (H_ * DH_) + h * DH_ + col] = acc * linv[row];
  }
}

// ---------------------------------------------------------------------------
// Kernel 3: MLP + residual + LayerNorm2 (fp32; <1% of total FLOPs)
__global__ __launch_bounds__(256) void mlp_kernel(const float* __restrict__ attv,
                                                  const float* __restrict__ slots,
                                                  const float* __restrict__ W1,
                                                  const float* __restrict__ b1,
                                                  const float* __restrict__ W2,
                                                  const float* __restrict__ b2,
                                                  const float* __restrict__ g2,
                                                  const float* __restrict__ beta2,
                                                  float* __restrict__ out) {
  const int row = blockIdx.x;               // b*64 + k
  const int tid = threadIdx.x;
  __shared__ float xin[DMLP_];
  __shared__ float hid[DMLP_];
  __shared__ float red[8];
  xin[tid]       = attv[(size_t)row * DMLP_ + tid];
  xin[tid + 256] = attv[(size_t)row * DMLP_ + tid + 256];
  __syncthreads();
  for (int c = tid; c < DMLP_; c += 256) {
    float acc = b1[c];
    for (int j = 0; j < DMLP_; ++j) acc = fmaf(xin[j], W1[(size_t)j * DMLP_ + c], acc);
    hid[c] = fmaxf(acc, 0.f);
  }
  __syncthreads();
  float acc = b2[tid];
  for (int j = 0; j < DMLP_; ++j) acc = fmaf(hid[j], W2[(size_t)j * DQ_ + tid], acc);
  const float y = acc + slots[(size_t)row * DQ_ + tid];
  const float mean = block_sum(y, red) * (1.f / DQ_);
  const float d = y - mean;
  const float var = block_sum(d * d, red) * (1.f / DQ_);
  out[(size_t)row * DQ_ + tid] = d * rsqrtf(var + 1e-5f) * g2[tid] + beta2[tid];
}

// ---------------------------------------------------------------------------
extern "C" void kernel_launch(void* const* d_in, const int* in_sizes, int n_in,
                              void* d_out, int out_size, void* d_ws, size_t ws_size,
                              hipStream_t stream) {
  const float* slots = (const float*)d_in[0];
  const float* hits  = (const float*)d_in[1];
  const float* ln1g  = (const float*)d_in[2];
  const float* ln1b  = (const float*)d_in[3];
  const float* Wq    = (const float*)d_in[4];
  const float* Wk    = (const float*)d_in[5];
  const float* Wv    = (const float*)d_in[6];
  const float* W1    = (const float*)d_in[7];
  const float* b1    = (const float*)d_in[8];
  const float* W2    = (const float*)d_in[9];
  const float* b2    = (const float*)d_in[10];
  const float* ln2g  = (const float*)d_in[11];
  const float* ln2b  = (const float*)d_in[12];
  float* out = (float*)d_out;

  char* ws = (char*)d_ws;
  bf16_t* wkb  = (bf16_t*)(ws + 0);                    // 256 KB
  bf16_t* wvb  = (bf16_t*)(ws + (256 << 10));          // 256 KB
  bf16_t* qws  = (bf16_t*)(ws + (512 << 10));          // 512 KB
  float*  attv = (float*) (ws + (1024 << 10));         // 1 MB
  float*  po   = (float*) (ws + (2048 << 10));         // 8 MB partial O
  float*  mst  = (float*) (ws + (10240 << 10));        // 128 KB partial m
  float*  lst  = (float*) (ws + (10368 << 10));        // 128 KB partial l

  pack_w_kernel<<<32, 256, 0, stream>>>(Wk, wkb);
  pack_w_kernel<<<32, 256, 0, stream>>>(Wv, wvb);
  ln_q_kernel<<<B_ * K_, 256, 0, stream>>>(slots, ln1g, ln1b, Wq, qws);
  attn_kernel<<<B_ * H_ * G_, 256, 0, stream>>>(hits, qws, wkb, wvb, po, mst, lst);
  attn_reduce_kernel<<<B_ * H_, 256, 0, stream>>>(po, mst, lst, attv);
  mlp_kernel<<<B_ * K_, 256, 0, stream>>>(attv, slots, W1, b1, W2, b2, ln2g, ln2b, out);
}
